// MultiHeadSelfAttention_17291538334455
// MI455X (gfx1250) — matrix operand order repaired and checked by reading, not compiled, not run on hardware
//
#include <hip/hip_runtime.h>
#include <hip/hip_bf16.h>

// ---------------------------------------------------------------------------
// MultiHeadSelfAttention (with faithful "full-head" attention bug) for gfx1250
//   qkv = x @ w_qkv^T ; q,k,v split ; scores = q@k^T * 1/8 ; softmax
//   out = probs @ v ; scramble reshape ; final = out2 @ w_proj^T + b_proj
// All GEMMs run as bf16 WMMA (v_wmma_f32_16x16x32_bf16), f32 accumulate,
// with an explicitly double-buffered K pipeline (unroll-by-2).
// ---------------------------------------------------------------------------

typedef __bf16 bf16;
typedef __attribute__((ext_vector_type(16))) __bf16 v16bf;
typedef __attribute__((ext_vector_type(8)))  __bf16 v8bf;
typedef __attribute__((ext_vector_type(8)))  float  v8f;

#define NB    4
#define S_LEN 2048
#define EMB   1024
#define HSZ   1024

__device__ __forceinline__ bf16 f2bf(float f) {
  unsigned u = __builtin_bit_cast(unsigned, f);
  unsigned r = u + 0x7fffu + ((u >> 16) & 1u);   // round-to-nearest-even
  unsigned short h = (unsigned short)(r >> 16);
  return __builtin_bit_cast(bf16, h);
}

__device__ __forceinline__ v8f zero8() {
  v8f z;
  #pragma unroll
  for (int i = 0; i < 8; ++i) z[i] = 0.0f;
  return z;
}

// ---------------- f32 -> bf16 conversion (grid-stride) ----------------------
__global__ __launch_bounds__(256)
void cvt_f32_bf16(const float* __restrict__ src, bf16* __restrict__ dst, int n) {
  int i = blockIdx.x * blockDim.x + threadIdx.x;
  int stride = gridDim.x * blockDim.x;
  for (; i < n; i += stride) dst[i] = f2bf(src[i]);
}

// ---------------- Generic NT bf16 WMMA GEMM ---------------------------------
// C[M,N] = A[M,K] (row-major, lda) x B[N,K]^T (row-major, ldb), f32 accum.
// Block: 256 thr = 8 waves as 4(M) x 2(N); block tile 128x128; wave tile 32x64.
enum { EPI_QKV = 0, EPI_SCORES = 1, EPI_SCRAMBLE = 2, EPI_FINAL = 3 };

// One fragment set: 2 A tiles (16x32) + 4 B tiles (32x16).
struct FragSet { v16bf a0, a1, b[4]; };

__device__ __forceinline__ void load_frags(const bf16* __restrict__ aPtr0,
                                           const bf16* __restrict__ aPtr1,
                                           const bf16* __restrict__ bPtr,
                                           int ldb, int k0, FragSet& f) {
  v8bf lo0 = *(const v8bf*)(aPtr0 + k0);
  v8bf hi0 = *(const v8bf*)(aPtr0 + k0 + 16);
  v8bf lo1 = *(const v8bf*)(aPtr1 + k0);
  v8bf hi1 = *(const v8bf*)(aPtr1 + k0 + 16);
  #pragma unroll
  for (int e = 0; e < 8; ++e) {
    f.a0[e] = lo0[e]; f.a0[8 + e] = hi0[e];
    f.a1[e] = lo1[e]; f.a1[8 + e] = hi1[e];
  }
  #pragma unroll
  for (int j = 0; j < 4; ++j) {
    v8bf blo = *(const v8bf*)(bPtr + (size_t)(16 * j) * ldb + k0);
    v8bf bhi = *(const v8bf*)(bPtr + (size_t)(16 * j) * ldb + k0 + 16);
    #pragma unroll
    for (int e = 0; e < 8; ++e) { f.b[j][e] = blo[e]; f.b[j][8 + e] = bhi[e]; }
  }
}

__device__ __forceinline__ void wmma_step(const FragSet& f, v8f acc[2][4]) {
  #pragma unroll
  for (int j = 0; j < 4; ++j) {
    acc[0][j] = __builtin_amdgcn_wmma_f32_16x16x32_bf16(
        false, f.a0, false, f.b[j], (short)0, acc[0][j], false, false);
    acc[1][j] = __builtin_amdgcn_wmma_f32_16x16x32_bf16(
        false, f.a1, false, f.b[j], (short)0, acc[1][j], false, false);
  }
}

template <int EPI>
__global__ __launch_bounds__(256)
void gemm_nt_wmma(const bf16* __restrict__ A, const bf16* __restrict__ B,
                  size_t aBatchStride, size_t bBatchStride,
                  int lda, int ldb, int K,
                  void* __restrict__ C0, void* __restrict__ C1,
                  void* __restrict__ C2,
                  const float* __restrict__ bias, float scale)
{
  const int lane = threadIdx.x & 31;
  const int wave = threadIdx.x >> 5;
  const int wm   = wave >> 1;                 // 0..3
  const int wn   = wave & 1;                  // 0..1
  const int b    = blockIdx.z;
  const int mW   = blockIdx.y * 128 + wm * 32;
  const int nW   = blockIdx.x * 128 + wn * 64;

  const int half = lane >> 4;                 // ISA 16-bit A/B lane-half
  const int l16  = lane & 15;

  const bf16* Ab = A + (size_t)b * aBatchStride;
  const bf16* Bb = B + (size_t)b * bBatchStride;

  // A fragment (16x32, MxK): lane row = l16; K halves at +half*8 and +16+half*8
  const bf16* aPtr0 = Ab + (size_t)(mW + l16) * lda + half * 8;
  const bf16* aPtr1 = aPtr0 + (size_t)16 * lda;
  // B fragment (32x16, KxN): lane column = l16; K halves at +half*8 and +16+half*8
  const bf16* bPtr  = Bb + (size_t)(nW + l16) * ldb + half * 8;

  v8f acc[2][4];
  #pragma unroll
  for (int i = 0; i < 2; ++i)
    #pragma unroll
    for (int j = 0; j < 4; ++j) acc[i][j] = zero8();

  // Double-buffered K pipeline; K is always a multiple of 64 here.
  FragSet f0, f1;
  load_frags(aPtr0, aPtr1, bPtr, ldb, 0,  f0);
  load_frags(aPtr0, aPtr1, bPtr, ldb, 32, f1);
  int k0 = 0;
  for (; k0 < K - 64; k0 += 64) {
    wmma_step(f0, acc);
    load_frags(aPtr0, aPtr1, bPtr, ldb, k0 + 64, f0);
    wmma_step(f1, acc);
    load_frags(aPtr0, aPtr1, bPtr, ldb, k0 + 96, f1);
  }
  wmma_step(f0, acc);
  wmma_step(f1, acc);

  // Epilogue: C/D f32 16x16 layout -> element (m = v + 8*half, n = l16)
  #pragma unroll
  for (int i = 0; i < 2; ++i) {
    #pragma unroll
    for (int j = 0; j < 4; ++j) {
      #pragma unroll
      for (int v = 0; v < 8; ++v) {
        const int m = mW + i * 16 + v + 8 * half;
        const int n = nW + j * 16 + l16;
        const float val = acc[i][j][v];
        if constexpr (EPI == EPI_QKV) {
          // n in [0,3072): split q / k / v; v stored TRANSPOSED for NT GEMM3
          bf16* q  = (bf16*)C0;
          bf16* kk = (bf16*)C1;
          bf16* vT = (bf16*)C2;
          const bf16 bv = f2bf(val);
          if (n < HSZ)
            q [(size_t)b * S_LEN * HSZ + (size_t)m * HSZ + n] = bv;
          else if (n < 2 * HSZ)
            kk[(size_t)b * S_LEN * HSZ + (size_t)m * HSZ + (n - HSZ)] = bv;
          else
            vT[(size_t)b * HSZ * S_LEN + (size_t)(n - 2 * HSZ) * S_LEN + m] = bv;
        } else if constexpr (EPI == EPI_SCORES) {
          float* sc = (float*)C0;
          sc[(size_t)b * S_LEN * S_LEN + (size_t)m * S_LEN + n] = val * scale;
        } else if constexpr (EPI == EPI_SCRAMBLE) {
          // out.swapaxes(1,2).reshape: dest flat index i = h*S + s
          bf16* o2 = (bf16*)C0;
          const unsigned idx = (unsigned)n * S_LEN + (unsigned)m;  // h*S + s
          const unsigned s2 = idx >> 10;        // / HSZ (1024)
          const unsigned h2 = idx & 1023u;      // % HSZ
          o2[(size_t)b * S_LEN * HSZ + (size_t)s2 * HSZ + h2] = f2bf(val);
        } else {  // EPI_FINAL
          float* o = (float*)C0;
          o[(size_t)b * S_LEN * EMB + (size_t)m * EMB + n] = val + bias[n];
        }
      }
    }
  }
}

// ---------------- Row softmax: scores f32 [B*S, S] -> probs bf16 ------------
__global__ __launch_bounds__(256)
void softmax_rows(const float* __restrict__ scores, bf16* __restrict__ probs) {
  __shared__ float red[256];
  const int row = blockIdx.x;                        // 0 .. B*S-1
  const float* src = scores + (size_t)row * S_LEN;
  bf16* dst = probs + (size_t)row * S_LEN;
  const int t = threadIdx.x;

  float vals[8];
  float mx = -3.4e38f;
  #pragma unroll
  for (int i = 0; i < 8; ++i) {
    vals[i] = src[t + 256 * i];
    mx = fmaxf(mx, vals[i]);
  }
  red[t] = mx; __syncthreads();
  for (int off = 128; off > 0; off >>= 1) {
    if (t < off) red[t] = fmaxf(red[t], red[t + off]);
    __syncthreads();
  }
  mx = red[0]; __syncthreads();

  float sum = 0.0f;
  #pragma unroll
  for (int i = 0; i < 8; ++i) { vals[i] = __expf(vals[i] - mx); sum += vals[i]; }
  red[t] = sum; __syncthreads();
  for (int off = 128; off > 0; off >>= 1) {
    if (t < off) red[t] += red[t + off];
    __syncthreads();
  }
  const float inv = 1.0f / red[0];
  #pragma unroll
  for (int i = 0; i < 8; ++i) dst[t + 256 * i] = f2bf(vals[i] * inv);
}

// ---------------------------------------------------------------------------
extern "C" void kernel_launch(void* const* d_in, const int* in_sizes, int n_in,
                              void* d_out, int out_size, void* d_ws, size_t ws_size,
                              hipStream_t stream) {
  (void)in_sizes; (void)n_in; (void)out_size; (void)ws_size;
  const float* x      = (const float*)d_in[0];   // [4,2048,1024]
  const float* w_qkv  = (const float*)d_in[1];   // [3072,1024]
  const float* w_proj = (const float*)d_in[2];   // [1024,1024]
  const float* b_proj = (const float*)d_in[3];   // [1024]
  float* out = (float*)d_out;                    // [4,2048,1024] f32

  // Workspace layout (256B aligned). out2 aliases x_bf16 (x dead after GEMM1).
  char* ws = (char*)d_ws;
  size_t o = 0;
  auto take = [&](size_t bytes) -> char* {
    char* p = ws + o;
    o += (bytes + 255) & ~(size_t)255;
    return p;
  };
  bf16*  xbf    = (bf16*)take((size_t)NB * S_LEN * EMB * 2);     // 16 MB
  bf16*  wqkvb  = (bf16*)take((size_t)3 * HSZ * EMB * 2);        //  6 MB
  bf16*  wprojb = (bf16*)take((size_t)EMB * HSZ * 2);            //  2 MB
  bf16*  qb     = (bf16*)take((size_t)NB * S_LEN * HSZ * 2);     // 16 MB
  bf16*  kb     = (bf16*)take((size_t)NB * S_LEN * HSZ * 2);     // 16 MB
  bf16*  vTb    = (bf16*)take((size_t)NB * HSZ * S_LEN * 2);     // 16 MB
  float* scores = (float*)take((size_t)NB * S_LEN * S_LEN * 4);  // 64 MB
  bf16*  probs  = (bf16*)take((size_t)NB * S_LEN * S_LEN * 2);   // 32 MB
  bf16*  out2   = xbf;                                           // alias

  const dim3 blk(256);

  // 0) convert operands to bf16
  cvt_f32_bf16<<<4096, blk, 0, stream>>>(x, xbf, NB * S_LEN * EMB);
  cvt_f32_bf16<<<2048, blk, 0, stream>>>(w_qkv, wqkvb, 3 * HSZ * EMB);
  cvt_f32_bf16<<<1024, blk, 0, stream>>>(w_proj, wprojb, EMB * HSZ);

  // 1) QKV: [S,E] @ w_qkv^T -> q,k (bf16) + v transposed (bf16)
  gemm_nt_wmma<EPI_QKV><<<dim3(3 * HSZ / 128, S_LEN / 128, NB), blk, 0, stream>>>(
      xbf, wqkvb, (size_t)S_LEN * EMB, 0, EMB, EMB, EMB,
      qb, kb, vTb, nullptr, 1.0f);

  // 2) scores = (q @ k^T) * 1/sqrt(64), f32
  gemm_nt_wmma<EPI_SCORES><<<dim3(S_LEN / 128, S_LEN / 128, NB), blk, 0, stream>>>(
      qb, kb, (size_t)S_LEN * HSZ, (size_t)S_LEN * HSZ, HSZ, HSZ, HSZ,
      scores, nullptr, nullptr, nullptr, 0.125f);

  // 3) row softmax -> probs bf16
  softmax_rows<<<NB * S_LEN, blk, 0, stream>>>(scores, probs);

  // 4) out = probs @ v  (v pre-transposed -> NT), scramble-reshape on store
  gemm_nt_wmma<EPI_SCRAMBLE><<<dim3(HSZ / 128, S_LEN / 128, NB), blk, 0, stream>>>(
      probs, vTb, (size_t)S_LEN * S_LEN, (size_t)HSZ * S_LEN, S_LEN, S_LEN, S_LEN,
      out2, nullptr, nullptr, nullptr, 1.0f);

  // 5) final = out2 @ w_proj^T + b_proj, f32
  gemm_nt_wmma<EPI_FINAL><<<dim3(EMB / 128, S_LEN / 128, NB), blk, 0, stream>>>(
      out2, wprojb, (size_t)S_LEN * HSZ, 0, HSZ, HSZ, HSZ,
      out, nullptr, nullptr, b_proj, 1.0f);
}
